// UltraSOTAModel_82549271429745
// MI455X (gfx1250) — matrix operand order repaired and checked by reading, not compiled, not run on hardware
//
#include <hip/hip_runtime.h>
#include <cstddef>
#include <cstdint>

// ---------------- model dims ----------------
constexpr int B_ = 8, S_ = 1024, D_ = 256, H_ = 8, HD_ = 32, WIN_ = 5, DST_ = 64, SPK_ = 153;

// ---------------- WMMA types (gfx1250, wave32) ----------------
typedef __attribute__((ext_vector_type(16))) __bf16 v16bf;
typedef __attribute__((ext_vector_type(8)))  float  v8f;

__device__ __forceinline__ v8f zero8() {
  v8f z;
#pragma unroll
  for (int i = 0; i < 8; ++i) z[i] = 0.0f;
  return z;
}

// raw (unconverted) fragment: 16 f32 in 4 float4 registers — lets all b128 loads
// of a K-step stay in flight concurrently instead of load->wait->cvt serialization.
struct raw16 { float4 x, y, z, w; };

__device__ __forceinline__ v16bf pk16(float4 a, float4 b, float4 c, float4 d) {
  v16bf f;
  f[0] = (__bf16)a.x;  f[1] = (__bf16)a.y;  f[2] = (__bf16)a.z;  f[3] = (__bf16)a.w;
  f[4] = (__bf16)b.x;  f[5] = (__bf16)b.y;  f[6] = (__bf16)b.z;  f[7] = (__bf16)b.w;
  f[8] = (__bf16)c.x;  f[9] = (__bf16)c.y;  f[10] = (__bf16)c.z; f[11] = (__bf16)c.w;
  f[12] = (__bf16)d.x; f[13] = (__bf16)d.y; f[14] = (__bf16)d.z; f[15] = (__bf16)d.w;
  return f;
}
__device__ __forceinline__ v16bf cvt16(const raw16& t) { return pk16(t.x, t.y, t.z, t.w); }

// A-matrix fragment raw load (16x32 tile, row-major, leading dim ld).
// lane<16 -> row=lane, K=0..15; lane>=16 -> row=lane-16, K=16..31
__device__ __forceinline__ raw16 ldrawA(const float* base, int ld) {
  int lane = threadIdx.x & 31;
  int r = lane & 15;
  int hk = (lane >> 4) << 4;  // 0 or 16
  const float* p = base + (size_t)r * ld + hk;
  raw16 t;
  t.x = *(const float4*)(p);
  t.y = *(const float4*)(p + 4);
  t.z = *(const float4*)(p + 8);
  t.w = *(const float4*)(p + 12);
  return t;
}
// B-matrix fragment raw load where B columns are ROWS of a row-major matrix (B = Wrows^T).
// lane<16 -> col=lane, K=0..15; lane>=16 -> col=lane-16, K=16..31
__device__ __forceinline__ raw16 ldrawB(const float* base, int ld) {
  int lane = threadIdx.x & 31;
  int n = lane & 15;
  int k0 = lane & 16;  // 0 or 16
  const float* p = base + (size_t)n * ld + k0;
  raw16 t;
  t.x = *(const float4*)(p);
  t.y = *(const float4*)(p + 4);
  t.z = *(const float4*)(p + 8);
  t.w = *(const float4*)(p + 12);
  return t;
}
__device__ __forceinline__ v16bf ldA32(const float* base, int ld) { return cvt16(ldrawA(base, ld)); }

// B-matrix fragment from a row-major f32 [32 x N] tile taken as-is (strided gather).
__device__ __forceinline__ v16bf ldBcols(const float* base, int ld) {
  int lane = threadIdx.x & 31;
  int n = lane & 15;
  int k0 = lane & 16;
  v16bf f;
#pragma unroll
  for (int i = 0; i < 16; ++i) f[i] = (__bf16)base[(size_t)(k0 + i) * ld + n];
  return f;
}

// ---------------- activations ----------------
__device__ __forceinline__ float gelu_erf(float x) { return 0.5f * x * (1.0f + erff(x * 0.7071067811865475f)); }
__device__ __forceinline__ float sigm(float x) { return 1.0f / (1.0f + __expf(-x)); }
template <int ACT>
__device__ __forceinline__ float act_fn(float x) {
  if constexpr (ACT == 1) return gelu_erf(x);
  else if constexpr (ACT == 2) return fmaxf(x, 0.0f);
  else if constexpr (ACT == 3) return x * sigm(x);
  else if constexpr (ACT == 4) return sigm(x);
  else return x;
}

// ---------------- WMMA GEMM: C[M,N] = act(A[M,K] @ W[N,K]^T + bias) (+resid) ----------------
// One wave per 16x64 output strip; software-pipelined: next K-step's 20 b128 loads are
// issued while the current 4 WMMAs execute.
template <int ACT>
__global__ __launch_bounds__(128) void gemm_bf16(const float* __restrict__ A, const float* __restrict__ W,
                                                 const float* __restrict__ bias, const float* __restrict__ resid,
                                                 float* __restrict__ C, int M, int N, int K) {
  int wid = blockIdx.x * 4 + (threadIdx.x >> 5);
  int nstrip = N >> 6;  // 64-wide strips (all N used are multiples of 64)
  int total = (M >> 4) * nstrip;
  if (wid >= total) return;  // wave-uniform -> EXEC all ones for WMMA
  int tm = wid / nstrip, tn = wid - tm * nstrip;
  const float* Ab = A + (size_t)tm * 16 * K;
  const float* Wb = W + (size_t)tn * 64 * K;
  v8f acc[4] = {zero8(), zero8(), zero8(), zero8()};
  raw16 ra = ldrawA(Ab, K);
  raw16 rb0 = ldrawB(Wb, K);
  raw16 rb1 = ldrawB(Wb + (size_t)16 * K, K);
  raw16 rb2 = ldrawB(Wb + (size_t)32 * K, K);
  raw16 rb3 = ldrawB(Wb + (size_t)48 * K, K);
  for (int k = 0; k < K; k += 32) {
    v16bf a = cvt16(ra);
    v16bf b0 = cvt16(rb0), b1 = cvt16(rb1), b2 = cvt16(rb2), b3 = cvt16(rb3);
    int kn = k + 32;
    if (kn < K) {  // prefetch next K-step (overlaps the 4 WMMAs below)
      ra = ldrawA(Ab + kn, K);
      rb0 = ldrawB(Wb + kn, K);
      rb1 = ldrawB(Wb + (size_t)16 * K + kn, K);
      rb2 = ldrawB(Wb + (size_t)32 * K + kn, K);
      rb3 = ldrawB(Wb + (size_t)48 * K + kn, K);
    }
    acc[0] = __builtin_amdgcn_wmma_f32_16x16x32_bf16(false, a, false, b0, (short)0, acc[0], false, false);
    acc[1] = __builtin_amdgcn_wmma_f32_16x16x32_bf16(false, a, false, b1, (short)0, acc[1], false, false);
    acc[2] = __builtin_amdgcn_wmma_f32_16x16x32_bf16(false, a, false, b2, (short)0, acc[2], false, false);
    acc[3] = __builtin_amdgcn_wmma_f32_16x16x32_bf16(false, a, false, b3, (short)0, acc[3], false, false);
  }
  int lane = threadIdx.x & 31;
  int col = lane & 15, rbase = (lane >> 4) << 3;
#pragma unroll
  for (int t = 0; t < 4; ++t) {
    int ncol = tn * 64 + t * 16 + col;
    float bv = bias ? bias[ncol] : 0.0f;
#pragma unroll
    for (int v = 0; v < 8; ++v) {
      int row = tm * 16 + rbase + v;
      float x = act_fn<ACT>(acc[t][v] + bv);
      size_t idx = (size_t)row * N + ncol;
      if (resid) x += resid[idx];
      C[idx] = x;
    }
  }
}

// ---------------- flash attention, one wave per (b,h,16-query tile) ----------------
// MASK==0: banned |q-k|<=WIN -> -1e9 ;  MASK==1: !keep[b,k] -> -1e30
template <int MASK>
__global__ __launch_bounds__(32) void attn_kernel(const float* __restrict__ Qb, const float* __restrict__ Kb,
                                                  const float* __restrict__ Vb, int ld,
                                                  const int* __restrict__ keep, float* __restrict__ AO) {
  int task = blockIdx.x;
  int QT = S_ / 16;
  int qt = task % QT; int t2 = task / QT; int h = t2 % H_; int b = t2 / H_;
  int lane = threadIdx.x;

  __shared__ float sc[16 * 32];
  __shared__ float mrow[16], lrow[16], srow[16];

  v16bf qf = ldA32(Qb + ((size_t)(b * S_ + qt * 16)) * ld + h * HD_, ld);
  v8f o0 = zero8(), o1 = zero8();
  if (lane < 16) { mrow[lane] = -3.0e38f; lrow[lane] = 0.0f; }
  __syncthreads();

  const float inv = 0.17677669529663687f;  // 1/sqrt(32)
  for (int kc = 0; kc < S_; kc += 32) {
    raw16 rk0 = ldrawB(Kb + ((size_t)(b * S_ + kc)) * ld + h * HD_, ld);
    raw16 rk1 = ldrawB(Kb + ((size_t)(b * S_ + kc + 16)) * ld + h * HD_, ld);
    v16bf kf0 = cvt16(rk0);
    v16bf kf1 = cvt16(rk1);
    v8f s0 = __builtin_amdgcn_wmma_f32_16x16x32_bf16(false, qf, false, kf0, (short)0, zero8(), false, false);
    v8f s1 = __builtin_amdgcn_wmma_f32_16x16x32_bf16(false, qf, false, kf1, (short)0, zero8(), false, false);
    // V fragments don't depend on the softmax -> issue their gathers now so they
    // overlap the LDS row-stats phase below.
    v16bf vf0 = ldBcols(Vb + ((size_t)(b * S_ + kc)) * ld + h * HD_, ld);
    v16bf vf1 = ldBcols(Vb + ((size_t)(b * S_ + kc)) * ld + h * HD_ + 16, ld);
    int col = lane & 15, rbase = (lane >> 4) << 3;
#pragma unroll
    for (int v = 0; v < 8; ++v) {
      sc[(rbase + v) * 32 + col] = s0[v];
      sc[(rbase + v) * 32 + 16 + col] = s1[v];
    }
    __syncthreads();
    if (lane < 16) {  // per-row online-softmax stats
      int r = lane, q = qt * 16 + r;
      float mold = mrow[r], mx = mold;
      for (int c = 0; c < 32; ++c) {
        float s = sc[r * 32 + c] * inv;
        if constexpr (MASK == 0) {
          int k = kc + c;
          if (abs(q - k) <= WIN_) s = -1.0e9f;
        } else {
          if (!keep[b * S_ + kc + c]) s = -1.0e30f;
        }
        sc[r * 32 + c] = s;
        mx = fmaxf(mx, s);
      }
      float fac = __expf(mold - mx);
      float sum = 0.0f;
      for (int c = 0; c < 32; ++c) {
        float p = __expf(sc[r * 32 + c] - mx);
        sc[r * 32 + c] = p;
        sum += p;
      }
      lrow[r] = lrow[r] * fac + sum;
      mrow[r] = mx;
      srow[r] = fac;
    }
    __syncthreads();
    v8f o0s, o1s;
#pragma unroll
    for (int v = 0; v < 8; ++v) {
      float f = srow[rbase + v];
      o0s[v] = o0[v] * f;
      o1s[v] = o1[v] * f;
    }
    v16bf pf = ldA32((const float*)sc, 32);
    o0 = __builtin_amdgcn_wmma_f32_16x16x32_bf16(false, pf, false, vf0, (short)0, o0s, false, false);
    o1 = __builtin_amdgcn_wmma_f32_16x16x32_bf16(false, pf, false, vf1, (short)0, o1s, false, false);
    __syncthreads();  // sc reusable next chunk (pf already in regs)
  }
  int col = lane & 15, rbase = (lane >> 4) << 3;
#pragma unroll
  for (int v = 0; v < 8; ++v) {
    int row = rbase + v, q = qt * 16 + row;
    float linv = 1.0f / fmaxf(lrow[row], 1e-30f);
    size_t o = ((size_t)(b * S_ + q)) * D_ + h * HD_;
    AO[o + col] = o0[v] * linv;
    AO[o + 16 + col] = o1[v] * linv;
  }
}

// ---------------- FFT real part: xr[b,k] = sum_n x[b,n] cos(2*pi*k*n/1024) ----------------
__global__ void fft_real_kernel(const float* __restrict__ x, float* __restrict__ xr) {
  __shared__ float xs[S_];
  int b = blockIdx.x, t = threadIdx.x;
  for (int i = t; i < S_; i += 256) xs[i] = x[b * S_ + i];
  __syncthreads();
  const float w = 6.283185307179586f / 1024.0f;
  for (int k = t; k < S_; k += 256) {
    float acc = 0.0f;
    for (int n = 0; n < S_; ++n) acc += xs[n] * __cosf(w * (float)((k * n) & 1023));
    xr[b * S_ + k] = acc;
  }
}

// ---------------- conv stack (BN eval factor = 1/sqrt(1+1e-5)) ----------------
#define BNF 0.9999950000374997f
__global__ void conv1_kernel(const float* xr, const float* w, const float* bias,
                             const float* g, const float* bb, float* out) {
  int idx = blockIdx.x * blockDim.x + threadIdx.x;
  if (idx >= B_ * 64 * S_) return;
  int s = idx & 1023, c = (idx >> 10) & 63, b = idx >> 16;
  float acc = bias[c];
#pragma unroll
  for (int t = 0; t < 7; ++t) {
    int ss = s + t - 3;
    if (ss >= 0 && ss < S_) acc += xr[b * S_ + ss] * w[c * 7 + t];
  }
  acc = gelu_erf(acc);
  out[idx] = acc * (g[c] * BNF) + bb[c];
}
__global__ void conv2_kernel(const float* h1, const float* w, const float* bias,
                             const float* g, const float* bb, float* out) {
  int idx = blockIdx.x * blockDim.x + threadIdx.x;
  if (idx >= B_ * 128 * S_) return;
  int s = idx & 1023, c = (idx >> 10) & 127, b = idx >> 17;
  float acc = bias[c];
  for (int ci = 0; ci < 64; ++ci)
#pragma unroll
    for (int t = 0; t < 5; ++t) {
      int ss = s + t - 2;
      if (ss >= 0 && ss < S_) acc += h1[(size_t)(b * 64 + ci) * S_ + ss] * w[(c * 64 + ci) * 5 + t];
    }
  acc = gelu_erf(acc);
  out[idx] = acc * (g[c] * BNF) + bb[c];
}
__global__ void conv3_kernel(const float* h2, const float* w, const float* bias,
                             const float* g, const float* bb, float* freq) {  // [B,S,D] output
  int idx = blockIdx.x * blockDim.x + threadIdx.x;
  if (idx >= B_ * 256 * S_) return;
  int s = idx & 1023, c = (idx >> 10) & 255, b = idx >> 18;
  float acc = bias[c];
  for (int ci = 0; ci < 128; ++ci)
#pragma unroll
    for (int t = 0; t < 3; ++t) {
      int ss = s + t - 1;
      if (ss >= 0 && ss < S_) acc += h2[(size_t)(b * 128 + ci) * S_ + ss] * w[(c * 128 + ci) * 3 + t];
    }
  freq[((size_t)(b * S_ + s)) * D_ + c] = acc * (g[c] * BNF) + bb[c];
}
__global__ void cat_kernel(const float* x, const float* freq, float* cat) {  // [B,S,512]
  int idx = blockIdx.x * blockDim.x + threadIdx.x;
  if (idx >= B_ * S_ * 512) return;
  int d = idx & 511, rest = idx >> 9;
  int s = rest & 1023, b = rest >> 10;
  cat[idx] = (d < 256) ? x[b * S_ + s] : freq[((size_t)(b * S_ + s)) * D_ + (d - 256)];
}

// ---------------- LayerNorm over 256 (+optional residual add) ----------------
__global__ void ln_kernel(const float* __restrict__ X, const float* g, const float* b,
                          const float* __restrict__ resid, float* __restrict__ out) {
  __shared__ float red[256];
  int row = blockIdx.x, t = threadIdx.x;
  size_t base = (size_t)row * 256;
  float x = X[base + t];
  red[t] = x; __syncthreads();
  for (int o = 128; o > 0; o >>= 1) { if (t < o) red[t] += red[t + o]; __syncthreads(); }
  float m = red[0] / 256.0f; __syncthreads();
  float d = x - m;
  red[t] = d * d; __syncthreads();
  for (int o = 128; o > 0; o >>= 1) { if (t < o) red[t] += red[t + o]; __syncthreads(); }
  float v = red[0] / 256.0f;
  float y = d * rsqrtf(v + 1e-5f) * g[t] + b[t];
  if (resid) y += resid[base + t];
  out[base + t] = y;
}

// ---------------- iterative top-k (SPK=153) per batch ----------------
__global__ void topk_kernel(const float* __restrict__ imp, int* __restrict__ keep) {
  __shared__ float vals[S_];
  __shared__ float rmax[256];
  __shared__ int ridx[256];
  int b = blockIdx.x, t = threadIdx.x;
  for (int i = t; i < S_; i += 256) { vals[i] = imp[b * S_ + i]; keep[b * S_ + i] = 0; }
  __syncthreads();
  for (int it = 0; it < SPK_; ++it) {
    float best = -3.0e38f; int bi = S_;
    for (int i = t; i < S_; i += 256) {
      float v = vals[i];
      if (v > best || (v == best && i < bi)) { best = v; bi = i; }
    }
    rmax[t] = best; ridx[t] = bi; __syncthreads();
    for (int o = 128; o > 0; o >>= 1) {
      if (t < o) {
        if (rmax[t + o] > rmax[t] || (rmax[t + o] == rmax[t] && ridx[t + o] < ridx[t])) {
          rmax[t] = rmax[t + o]; ridx[t] = ridx[t + o];
        }
      }
      __syncthreads();
    }
    if (t == 0) { keep[b * S_ + ridx[0]] = 1; vals[ridx[0]] = -3.0e38f; }
    __syncthreads();
  }
}

// ---------------- sequential SSM scan: h_t = h_{t-1} @ A^T + u_t ----------------
__global__ void ssm_scan_kernel(const float* __restrict__ A, const float* __restrict__ u,
                                float* __restrict__ hs) {
  __shared__ float Am[DST_ * DST_];
  __shared__ float h[DST_];
  int b = blockIdx.x, t = threadIdx.x;  // 64 threads
  for (int i = t; i < DST_ * DST_; i += DST_) Am[i] = A[i];
  h[t] = 0.0f;
  __syncthreads();
  for (int st = 0; st < S_; ++st) {
    float acc = u[((size_t)b * S_ + st) * DST_ + t];
    for (int j = 0; j < DST_; ++j) acc += Am[t * DST_ + j] * h[j];
    __syncthreads();
    h[t] = acc;
    hs[((size_t)b * S_ + st) * DST_ + t] = acc;
    __syncthreads();
  }
}

// ---------------- elementwise helpers ----------------
__global__ void ew_mul_kernel(float* a, const float* b, int n) {
  int i = blockIdx.x * blockDim.x + threadIdx.x;
  if (i < n) a[i] *= b[i];
}
__global__ void ew_gate_add_kernel(const float* y, const float* gate, const float* sp, float* out, int n) {
  int i = blockIdx.x * blockDim.x + threadIdx.x;
  if (i < n) out[i] = y[i] * gate[i] + sp[i];
}
__global__ void mean_over_s_kernel(const float* __restrict__ X, float* outA, float* outB, int S, int D) {
  int idx = blockIdx.x * blockDim.x + threadIdx.x;
  if (idx >= B_ * D) return;
  int b = idx / D, d = idx - b * D;
  float acc = 0.0f;
  for (int s = 0; s < S; ++s) acc += X[((size_t)b * S + s) * D + d];
  acc /= (float)S;
  if (outA) outA[idx] = acc;
  if (outB) outB[idx] = acc;
}
template <int ACT>
__global__ void dense_small(const float* __restrict__ A, const float* __restrict__ W,
                            const float* __restrict__ bias, float* __restrict__ C, int M, int N, int K) {
  int idx = blockIdx.x * blockDim.x + threadIdx.x;
  if (idx >= M * N) return;
  int m = idx / N, n = idx - m * N;
  float acc = bias ? bias[n] : 0.0f;
  for (int k = 0; k < K; ++k) acc += A[(size_t)m * K + k] * W[(size_t)n * K + k];
  C[idx] = act_fn<ACT>(acc);
}
__global__ void l2norm_kernel(const float* Z, float* out) {  // grid B_, block 64
  __shared__ float red[64];
  int b = blockIdx.x, t = threadIdx.x;
  float z = Z[b * 64 + t];
  red[t] = z * z; __syncthreads();
  for (int o = 32; o > 0; o >>= 1) { if (t < o) red[t] += red[t + o]; __syncthreads(); }
  out[b * 64 + t] = z / fmaxf(sqrtf(red[0]), 1e-12f);
}

// ---------------- host orchestration ----------------
static inline int cdiv(int a, int b) { return (a + b - 1) / b; }

template <int ACT>
static void gemm_go(hipStream_t st, const float* A, const float* W, const float* bias,
                    const float* resid, float* C, int M, int N, int K) {
  int tiles = (M / 16) * (N / 64);
  gemm_bf16<ACT><<<cdiv(tiles, 4), 128, 0, st>>>(A, W, bias, resid, C, M, N, K);
}

// input indices (setup_inputs dict insertion order, recursively flattened; lin={w,b}, lnp={g,b})
enum {
  IX_X = 0,
  IX_C1W, IX_C1B, IX_BN1G, IX_BN1B, IX_C2W, IX_C2B, IX_BN2G, IX_BN2B,
  IX_C3W, IX_C3B, IX_BN3G, IX_BN3B, IX_F1W, IX_F1B, IX_F2W, IX_F2B, IX_LNG, IX_LNB,   // freq ..18
  IX_M1W, IX_M1B, IX_M2W, IX_M2B, IX_QW, IX_QB, IX_KW, IX_KB, IX_VW, IX_VB, IX_OW, IX_OB,  // sub ..30
  IX_G1W, IX_G1B, IX_G2W, IX_G2B, IX_INW, IX_INB, IX_SOW, IX_SOB,                     // sp ..38
  IX_SA, IX_SBW, IX_SBB, IX_SCW, IX_SCB, IX_DMW, IX_DMB, IX_S1W, IX_S1B, IX_S2W, IX_S2B, IX_SGW, IX_SGB, // ssm ..51
  IX_FUS = 52,  // 3 blocks x (l1.w,l1.b,l2.w,l2.b,ln.g,ln.b)
  IX_REC = 70, IX_SER = 76, IX_PT = 82, IX_CON = 86
};

// workspace layout (floats); total ~25.2M floats (~101 MB)
constexpr size_t OFF_XR = 0, OFF_IMP = 8192, OFF_KEEP = 16384, OFF_GV = 24576;
constexpr size_t OFF_HT0 = 26624, OFF_HT1 = 30720, OFF_HT2 = 32768, OFF_HT3 = 33792;
constexpr size_t BIG = 2097152;  // one [B,S,D] f32 tensor
constexpr size_t OFF_A = 40960;           // feats
constexpr size_t OFF_B = OFF_A + BIG;     // sub
constexpr size_t OFF_C = OFF_B + BIG;     // sp
constexpr size_t OFF_D = OFF_C + BIG;     // fused
constexpr size_t OFF_E = OFF_D + BIG;     // freq / xm / AO2 / pt1 scratch
constexpr size_t OFF_F = OFF_E + BIG;     // f2out / AO1 / y scratch
constexpr size_t OFF_G2 = OFF_F + BIG;    // gate
constexpr size_t OFF_H = OFF_G2 + BIG;    // 4.19M: cat512 / m1out / ssm smalls / fus l1 / pt2(+I)
constexpr size_t OFF_I = OFF_H + 4194304; // 6.29M: h1+h2 / QKV / qkv768

extern "C" void kernel_launch(void* const* d_in, const int* in_sizes, int n_in,
                              void* d_out, int out_size, void* d_ws, size_t ws_size,
                              hipStream_t stream) {
  (void)in_sizes; (void)n_in; (void)out_size; (void)ws_size;
  auto IN = [&](int i) { return (const float*)d_in[i]; };
  float* ws = (float*)d_ws;
  float* out = (float*)d_out;

  float* xr = ws + OFF_XR;
  float* imp = ws + OFF_IMP;
  int* keep = (int*)(ws + OFF_KEEP);
  float* gv = ws + OFF_GV;
  float* feats = ws + OFF_A;
  float* subb = ws + OFF_B;
  float* spb = ws + OFF_C;
  float* fused = ws + OFF_D;
  float* bufE = ws + OFF_E;
  float* bufF = ws + OFF_F;
  float* gate = ws + OFF_G2;
  float* bufH = ws + OFF_H;
  float* bufI = ws + OFF_I;
  // ssm smalls inside H
  float* t128 = bufH;
  float* sel = bufH + 1048576;
  float* ubuf = bufH + 1572864;
  float* hsb = bufH + 2097152;
  float* timp = bufH + 2621440;

  const int NROW = B_ * S_;  // 8192

  // ---- FrequencyAugmentedModule ----
  fft_real_kernel<<<B_, 256, 0, stream>>>(IN(IX_X), xr);
  conv1_kernel<<<cdiv(B_ * 64 * S_, 256), 256, 0, stream>>>(xr, IN(IX_C1W), IN(IX_C1B), IN(IX_BN1G), IN(IX_BN1B), bufI);
  conv2_kernel<<<cdiv(B_ * 128 * S_, 256), 256, 0, stream>>>(bufI, IN(IX_C2W), IN(IX_C2B), IN(IX_BN2G), IN(IX_BN2B), bufI + 524288);
  conv3_kernel<<<cdiv(B_ * 256 * S_, 256), 256, 0, stream>>>(bufI + 524288, IN(IX_C3W), IN(IX_C3B), IN(IX_BN3G), IN(IX_BN3B), bufE);
  cat_kernel<<<cdiv(B_ * S_ * 512, 256), 256, 0, stream>>>(IN(IX_X), bufE, bufH);
  gemm_go<1>(stream, bufH, IN(IX_F1W), IN(IX_F1B), nullptr, bufE, NROW, 256, 512);  // f1 + gelu
  gemm_go<0>(stream, bufE, IN(IX_F2W), IN(IX_F2B), nullptr, bufF, NROW, 256, 256);  // f2
  ln_kernel<<<NROW, 256, 0, stream>>>(bufF, IN(IX_LNG), IN(IX_LNB), nullptr, feats);

  // ---- SubAdjacentAttention + residual ----
  gemm_go<2>(stream, feats, IN(IX_M1W), IN(IX_M1B), nullptr, bufH, NROW, 512, 256);  // m1 relu
  gemm_go<0>(stream, bufH, IN(IX_M2W), IN(IX_M2B), nullptr, bufE, NROW, 256, 512);   // m2 -> xm
  gemm_go<0>(stream, bufE, IN(IX_QW), IN(IX_QB), nullptr, bufI, NROW, 256, 256);
  gemm_go<0>(stream, bufE, IN(IX_KW), IN(IX_KB), nullptr, bufI + BIG, NROW, 256, 256);
  gemm_go<0>(stream, bufE, IN(IX_VW), IN(IX_VB), nullptr, bufI + 2 * BIG, NROW, 256, 256);
  attn_kernel<0><<<B_ * H_ * (S_ / 16), 32, 0, stream>>>(bufI, bufI + BIG, bufI + 2 * BIG, 256, nullptr, bufF);
  gemm_go<0>(stream, bufF, IN(IX_OW), IN(IX_OB), feats, subb, NROW, 256, 256);  // o-proj + resid

  // ---- SparseAttentionModule + residual ----
  gemm_go<2>(stream, subb, IN(IX_G1W), IN(IX_G1B), nullptr, timp, NROW, 64, 256);
  dense_small<4><<<cdiv(NROW, 256), 256, 0, stream>>>(timp, IN(IX_G2W), IN(IX_G2B), imp, NROW, 1, 64);
  topk_kernel<<<B_, 256, 0, stream>>>(imp, keep);
  gemm_go<0>(stream, subb, IN(IX_INW), IN(IX_INB), nullptr, bufI, NROW, 768, 256);  // qkv
  attn_kernel<1><<<B_ * H_ * (S_ / 16), 32, 0, stream>>>(bufI, bufI + 256, bufI + 512, 768, keep, bufE);
  gemm_go<0>(stream, bufE, IN(IX_SOW), IN(IX_SOB), subb, spb, NROW, 256, 256);

  // ---- MambaLikeSSM ----
  gemm_go<3>(stream, spb, IN(IX_S1W), IN(IX_S1B), nullptr, t128, NROW, 128, 256);   // s1 silu
  gemm_go<4>(stream, t128, IN(IX_S2W), IN(IX_S2B), nullptr, sel, NROW, 64, 128);    // s2 sigmoid
  gemm_go<0>(stream, spb, IN(IX_SBW), IN(IX_SBB), nullptr, ubuf, NROW, 64, 256);    // B(x)
  ew_mul_kernel<<<cdiv(NROW * 64, 256), 256, 0, stream>>>(ubuf, sel, NROW * 64);    // u = B(x)*sel
  ssm_scan_kernel<<<B_, 64, 0, stream>>>(IN(IX_SA), ubuf, hsb);
  gemm_go<0>(stream, hsb, IN(IX_SCW), IN(IX_SCB), nullptr, bufF, NROW, 256, 64);    // C(hs)
  gemm_go<0>(stream, spb, IN(IX_DMW), IN(IX_DMB), bufF, bufF, NROW, 256, 256);      // + Dm(sp)
  gemm_go<4>(stream, spb, IN(IX_SGW), IN(IX_SGB), nullptr, gate, NROW, 256, 256);   // sigmoid(g(sp))
  ew_gate_add_kernel<<<cdiv(NROW * 256, 256), 256, 0, stream>>>(bufF, gate, spb, fused, NROW * 256);

  // ---- 3 fusion MLP blocks with LN + residual ----
  for (int i = 0; i < 3; ++i) {
    int base = IX_FUS + 6 * i;
    gemm_go<1>(stream, fused, IN(base + 0), IN(base + 1), nullptr, bufH, NROW, 512, 256);
    gemm_go<0>(stream, bufH, IN(base + 2), IN(base + 3), nullptr, bufE, NROW, 256, 512);
    ln_kernel<<<NROW, 256, 0, stream>>>(bufE, IN(base + 4), IN(base + 5), fused, fused);
  }

  // ---- heads ----
  // g = fused.mean(axis=1)  (also written to d_out g slot)
  mean_over_s_kernel<<<cdiv(B_ * 256, 256), 256, 0, stream>>>(fused, gv, out + 16904, S_, 256);
  // recon
  dense_small<1><<<cdiv(8 * 512, 256), 256, 0, stream>>>(gv, IN(IX_REC + 0), IN(IX_REC + 1), ws + OFF_HT0, 8, 512, 256);
  dense_small<1><<<cdiv(8 * 256, 256), 256, 0, stream>>>(ws + OFF_HT0, IN(IX_REC + 2), IN(IX_REC + 3), ws + OFF_HT1, 8, 256, 512);
  dense_small<0><<<cdiv(8 * 1024, 256), 256, 0, stream>>>(ws + OFF_HT1, IN(IX_REC + 4), IN(IX_REC + 5), out + 0, 8, 1024, 256);
  // series
  dense_small<2><<<cdiv(8 * 128, 256), 256, 0, stream>>>(gv, IN(IX_SER + 0), IN(IX_SER + 1), ws + OFF_HT2, 8, 128, 256);
  dense_small<2><<<1, 256, 0, stream>>>(ws + OFF_HT2, IN(IX_SER + 2), IN(IX_SER + 3), ws + OFF_HT3, 8, 64, 128);
  dense_small<4><<<1, 256, 0, stream>>>(ws + OFF_HT3, IN(IX_SER + 4), IN(IX_SER + 5), out + 8192, 8, 1, 64);
  // point: sigmoid(pt.l2(relu(pt.l1(fused)))).mean(axis=1)
  gemm_go<2>(stream, fused, IN(IX_PT + 0), IN(IX_PT + 1), nullptr, bufE, NROW, 128, 256);
  gemm_go<4>(stream, bufE, IN(IX_PT + 2), IN(IX_PT + 3), nullptr, bufH, NROW, 1024, 128);  // pt2 spans H..I
  mean_over_s_kernel<<<cdiv(B_ * 1024, 256), 256, 0, stream>>>(bufH, out + 8200, nullptr, S_, 1024);
  // contr
  dense_small<2><<<cdiv(8 * 128, 256), 256, 0, stream>>>(gv, IN(IX_CON + 0), IN(IX_CON + 1), ws + OFF_HT2, 8, 128, 256);
  dense_small<0><<<1, 256, 0, stream>>>(ws + OFF_HT2, IN(IX_CON + 2), IN(IX_CON + 3), ws + OFF_HT0, 8, 64, 128);
  l2norm_kernel<<<B_, 64, 0, stream>>>(ws + OFF_HT0, out + 16392);
}
